// CausalMultiHeadAttention_66666482368602
// MI455X (gfx1250) — matrix operand order repaired and checked by reading, not compiled, not run on hardware
//
#include <hip/hip_runtime.h>
#include <hip/hip_bf16.h>

typedef __attribute__((ext_vector_type(16))) _Float16 v16h;
typedef __attribute__((ext_vector_type(8)))  _Float16 v8h;
typedef __attribute__((ext_vector_type(4)))  _Float16 v4h;
typedef __attribute__((ext_vector_type(8)))  float    v8f;

#define D_MODEL 1024
#define SEQ     2048
#define NB      2
#define NH      16
#define DK      64
#define MTOT    (NB * SEQ)   // 4096

static __device__ __forceinline__ v16h cat16(v8h lo, v8h hi) {
    return __builtin_shufflevector(lo, hi, 0,1,2,3,4,5,6,7,8,9,10,11,12,13,14,15);
}

// A-operand 16x32 f16 tile. base -> element (row0, k0). lane l (<16): row l, K {0..7,16..23};
// lane l (>=16): row l-16, K {8..15,24..31}.  (ISA 7.12.2, 16-bit A 16x32)
static __device__ __forceinline__ v16h load_a16x32(const _Float16* base, int rowStride, int lane) {
    const _Float16* p = base + (lane & 15) * rowStride + ((lane & 16) ? 8 : 0);
    v8h lo = *(const v8h*)(p);
    v8h hi = *(const v8h*)(p + 16);
    return cat16(lo, hi);
}

// B-operand 32x16 f16 tile. base -> element (k0, n0). lane l holds column N=l%16, K {0..7,16..23} (l<16) or {8..15,24..31} (l>=16).
static __device__ __forceinline__ v16h load_b32x16(const _Float16* base, int rowStride, int lane) {
    const _Float16* p = base + (lane & 15) + ((lane & 16) ? 8 : 0) * rowStride;
    v16h r;
#pragma unroll
    for (int s = 0; s < 8; ++s) {
        r[s]     = p[s * rowStride];
        r[8 + s] = p[(16 + s) * rowStride];
    }
    return r;
}

static __device__ __forceinline__ v8f wmma_f16(v16h a, v16h b, v8f c) {
    return __builtin_amdgcn_wmma_f32_16x16x32_f16(false, a, false, b, (short)0, c, false, false);
}

// ---------------- prep kernels ----------------
// vectorized f32 -> f16 (4 elems/thread: b128 load, b64 store)
__global__ __launch_bounds__(256) void cvt_x_kernel(const float* __restrict__ src,
                                                    _Float16* __restrict__ dst, int n4) {
    int i = blockIdx.x * 256 + threadIdx.x;
    if (i < n4) {
        float4 f = ((const float4*)src)[i];
        v4h h = { (_Float16)f.x, (_Float16)f.y, (_Float16)f.z, (_Float16)f.w };
        ((v4h*)dst)[i] = h;
    }
}

// W is [O][D] row-major f32; write Wt [D][O] f16 so GEMM B = W^T has contiguous rows.
__global__ __launch_bounds__(256) void transpose_w_kernel(const float* __restrict__ src,
                                                          _Float16* __restrict__ dst) {
    int i = blockIdx.x * 256 + threadIdx.x;      // over 1024*1024
    int o = i >> 10;
    int d = i & 1023;
    dst[d * D_MODEL + o] = (_Float16)src[i];
}

// ---------------- fused QKV + RoPE ----------------
// grid: (48, 32) blocks x 128 threads. blockIdx.x = sel*16 + n64tile.
// Each wave computes a 32x64 output tile (2 M-tiles x 4 N-tiles, 8 WMMA accs)
// so every B operand is reused by two WMMAs -> 2x arithmetic intensity.
__global__ __launch_bounds__(128) void qkv_rope_kernel(const _Float16* __restrict__ xh,
                                                       const _Float16* __restrict__ wqt,
                                                       const _Float16* __restrict__ wkt,
                                                       const _Float16* __restrict__ wvt,
                                                       const int* __restrict__ tpos,
                                                       _Float16* __restrict__ Qh,   // [B][H][S][DK]
                                                       _Float16* __restrict__ Kt,   // [B][H][DK][S]
                                                       _Float16* __restrict__ Vh) { // [B][H][S][DK]
    const int lane = threadIdx.x & 31;
    const int w    = threadIdx.x >> 5;
    const int sel  = blockIdx.x >> 4;            // 0=q 1=k 2=v
    const int n0   = (blockIdx.x & 15) * 64;
    const int m0   = (blockIdx.y * 4 + w) * 32;

    const _Float16* wt = (sel == 0) ? wqt : (sel == 1) ? wkt : wvt;

    v8f acc[2][4] = {};
    for (int kk = 0; kk < D_MODEL; kk += 32) {
        v16h av0 = load_a16x32(xh + (m0 +  0) * D_MODEL + kk, D_MODEL, lane);
        v16h av1 = load_a16x32(xh + (m0 + 16) * D_MODEL + kk, D_MODEL, lane);
#pragma unroll
        for (int t = 0; t < 4; ++t) {
            v16h bv = load_b32x16(wt + kk * D_MODEL + n0 + t * 16, D_MODEL, lane);
            acc[0][t] = wmma_f16(av0, bv, acc[0][t]);
            acc[1][t] = wmma_f16(av1, bv, acc[1][t]);
        }
    }

    const int half8 = (lane >> 4) << 3;
#pragma unroll
    for (int mi = 0; mi < 2; ++mi) {
#pragma unroll
        for (int t = 0; t < 4; ++t) {
#pragma unroll
            for (int r = 0; r < 8; ++r) {
                float val = acc[mi][t][r];
                int m  = m0 + mi * 16 + r + half8;
                int n  = n0 + t * 16 + (lane & 15);
                int bb = m >> 11;                    // / SEQ
                int s  = m & (SEQ - 1);
                int h  = n >> 6;                     // / DK
                int dd = n & (DK - 1);
                if (sel == 2) {
                    Vh[(((bb * NH + h) * SEQ) + s) * DK + dd] = (_Float16)val;
                } else {
                    // RoPE: pair exchange is lane^1 (adjacent n, same row, same VGPR)
                    float other = __shfl_xor(val, 1, 32);
                    float pos   = (float)tpos[s];
                    float freq  = __expf(-(float)(dd & ~1) * 0.14391156516f); // ln(1e4)/64
                    float ang   = pos * freq;
                    float cs = cosf(ang), sn = sinf(ang);
                    float rot = val * cs + other * ((dd & 1) ? sn : -sn);
                    if (sel == 0)
                        Qh[(((bb * NH + h) * SEQ) + s) * DK + dd] = (_Float16)rot;
                    else
                        Kt[(((bb * NH + h) * DK) + dd) * SEQ + s] = (_Float16)rot;
                }
            }
        }
    }
}

// ---------------- flash causal attention ----------------
// grid: 1024 blocks x 128 threads; each wave owns one (b,h,16-query tile).
__global__ __launch_bounds__(128) void attn_kernel(const _Float16* __restrict__ Qh,
                                                   const _Float16* __restrict__ Kt,
                                                   const _Float16* __restrict__ Vh,
                                                   _Float16* __restrict__ ctxh) { // [B*S][D_MODEL]
    __shared__ __align__(16) _Float16 ldsP[4][16 * 32];

    const int lane = threadIdx.x & 31;
    const int w    = threadIdx.x >> 5;
    const int wg   = blockIdx.x * 4 + w;
    const int qt   = wg & 127;
    const int h    = (wg >> 7) & (NH - 1);
    const int bb   = wg >> 11;
    const int q0   = qt * 16;

    const _Float16* qbase = Qh + (((bb * NH + h) * SEQ) + q0) * DK;
    const _Float16* ktb   = Kt + ((bb * NH + h) * DK) * SEQ;
    const _Float16* vb    = Vh + ((bb * NH + h) * SEQ) * DK;

    const v16h qa0 = load_a16x32(qbase + 0,  DK, lane);
    const v16h qa1 = load_a16x32(qbase + 32, DK, lane);

    const int half8 = (lane >> 4) << 3;
    float m_i[8], l_i[8], alpha[8];
#pragma unroll
    for (int r = 0; r < 8; ++r) { m_i[r] = -1e30f; l_i[r] = 0.0f; }
    v8f o[4] = {};

    for (int kb = 0; kb < q0 + 16; kb += 32) {
        v8f sc[2] = {};
#pragma unroll
        for (int t = 0; t < 2; ++t) {
            v16h b0 = load_b32x16(ktb + 0  * SEQ + kb + t * 16, SEQ, lane);
            sc[t] = wmma_f16(qa0, b0, sc[t]);
            v16h b1 = load_b32x16(ktb + 32 * SEQ + kb + t * 16, SEQ, lane);
            sc[t] = wmma_f16(qa1, b1, sc[t]);
        }
        // online softmax over these 32 keys; stats per (r, lane-half) == per query row
#pragma unroll
        for (int r = 0; r < 8; ++r) {
            float s0 = sc[0][r] * 0.125f;        // 1/sqrt(64)
            float s1 = sc[1][r] * 0.125f;
            int qr = q0 + r + half8;
            int kc = kb + (lane & 15);
            if (kc      > qr) s0 = -1e30f;
            if (kc + 16 > qr) s1 = -1e30f;
            float cm = fmaxf(s0, s1);
#pragma unroll
            for (int off = 1; off < 16; off <<= 1) cm = fmaxf(cm, __shfl_xor(cm, off, 32));
            float mn = fmaxf(m_i[r], cm);
            float p0 = __expf(s0 - mn), p1 = __expf(s1 - mn);
            float rs = p0 + p1;
#pragma unroll
            for (int off = 1; off < 16; off <<= 1) rs += __shfl_xor(rs, off, 32);
            float a = __expf(m_i[r] - mn);
            l_i[r] = l_i[r] * a + rs;
            m_i[r] = mn;
            alpha[r] = a;
            int qq = r + half8;
            ldsP[w][qq * 32 +      (lane & 15)] = (_Float16)p0;
            ldsP[w][qq * 32 + 16 + (lane & 15)] = (_Float16)p1;
        }
#pragma unroll
        for (int t = 0; t < 4; ++t)
#pragma unroll
            for (int r = 0; r < 8; ++r) o[t][r] *= alpha[r];

        // re-layout P (D-layout) -> A-layout via per-wave LDS (in-order within wave)
        const _Float16* pb = &ldsP[w][(lane & 15) * 32 + ((lane & 16) ? 8 : 0)];
        v8h lo = *(const v8h*)pb;
        v8h hi = *(const v8h*)(pb + 16);
        v16h pa = cat16(lo, hi);
#pragma unroll
        for (int t2 = 0; t2 < 4; ++t2) {
            v16h bv = load_b32x16(vb + kb * DK + t2 * 16, DK, lane);
            o[t2] = wmma_f16(pa, bv, o[t2]);
        }
    }

    float inv[8];
#pragma unroll
    for (int r = 0; r < 8; ++r) inv[r] = 1.0f / l_i[r];
#pragma unroll
    for (int t2 = 0; t2 < 4; ++t2)
#pragma unroll
        for (int r = 0; r < 8; ++r) {
            int s = q0 + r + half8;
            int n = h * DK + t2 * 16 + (lane & 15);
            ctxh[((bb * SEQ) + s) * D_MODEL + n] = (_Float16)(o[t2][r] * inv[r]);
        }
}

// ---------------- output projection ----------------
// grid: (16, 32) x 128 threads; wave computes 32x64 of out = ctx * Wo^T (f32 out)
__global__ __launch_bounds__(128) void proj_kernel(const _Float16* __restrict__ ctxh,
                                                   const _Float16* __restrict__ wot,
                                                   float* __restrict__ out) {
    const int lane = threadIdx.x & 31;
    const int w    = threadIdx.x >> 5;
    const int n0   = blockIdx.x * 64;
    const int m0   = (blockIdx.y * 4 + w) * 32;

    v8f acc[2][4] = {};
    for (int kk = 0; kk < D_MODEL; kk += 32) {
        v16h av0 = load_a16x32(ctxh + (m0 +  0) * D_MODEL + kk, D_MODEL, lane);
        v16h av1 = load_a16x32(ctxh + (m0 + 16) * D_MODEL + kk, D_MODEL, lane);
#pragma unroll
        for (int t = 0; t < 4; ++t) {
            v16h bv = load_b32x16(wot + kk * D_MODEL + n0 + t * 16, D_MODEL, lane);
            acc[0][t] = wmma_f16(av0, bv, acc[0][t]);
            acc[1][t] = wmma_f16(av1, bv, acc[1][t]);
        }
    }
    const int half8 = (lane >> 4) << 3;
#pragma unroll
    for (int mi = 0; mi < 2; ++mi)
#pragma unroll
        for (int t = 0; t < 4; ++t)
#pragma unroll
            for (int r = 0; r < 8; ++r) {
                int m = m0 + mi * 16 + r + half8;
                int n = n0 + t * 16 + (lane & 15);
                out[m * D_MODEL + n] = acc[mi][t][r];
            }
}

extern "C" void kernel_launch(void* const* d_in, const int* in_sizes, int n_in,
                              void* d_out, int out_size, void* d_ws, size_t ws_size,
                              hipStream_t stream) {
    const float* x    = (const float*)d_in[0];
    const int*   tpos = (const int*)d_in[1];
    const float* wq   = (const float*)d_in[2];
    const float* wk   = (const float*)d_in[3];
    const float* wv   = (const float*)d_in[4];
    const float* wo   = (const float*)d_in[5];
    float* out = (float*)d_out;

    char* ws = (char*)d_ws;
    const size_t SZ_X  = (size_t)MTOT * D_MODEL * 2;      // 8 MB
    const size_t SZ_W  = (size_t)D_MODEL * D_MODEL * 2;   // 2 MB
    const size_t SZ_T  = (size_t)NB * NH * SEQ * DK * 2;  // 8 MB
    _Float16* xh   = (_Float16*)(ws);
    _Float16* wqt  = (_Float16*)(ws + SZ_X);
    _Float16* wkt  = (_Float16*)(ws + SZ_X + SZ_W);
    _Float16* wvt  = (_Float16*)(ws + SZ_X + 2 * SZ_W);
    _Float16* wot  = (_Float16*)(ws + SZ_X + 3 * SZ_W);
    _Float16* Qh   = (_Float16*)(ws + SZ_X + 4 * SZ_W);
    _Float16* Kt   = (_Float16*)(ws + SZ_X + 4 * SZ_W + SZ_T);
    _Float16* Vh   = (_Float16*)(ws + SZ_X + 4 * SZ_W + 2 * SZ_T);
    _Float16* ctxh = (_Float16*)(ws + SZ_X + 4 * SZ_W + 3 * SZ_T);

    const int NX4 = (MTOT * D_MODEL) / 4;                 // 1048576
    cvt_x_kernel<<<NX4 / 256, 256, 0, stream>>>(x, xh, NX4);
    transpose_w_kernel<<<(D_MODEL * D_MODEL) / 256, 256, 0, stream>>>(wq, wqt);
    transpose_w_kernel<<<(D_MODEL * D_MODEL) / 256, 256, 0, stream>>>(wk, wkt);
    transpose_w_kernel<<<(D_MODEL * D_MODEL) / 256, 256, 0, stream>>>(wv, wvt);
    transpose_w_kernel<<<(D_MODEL * D_MODEL) / 256, 256, 0, stream>>>(wo, wot);

    qkv_rope_kernel<<<dim3(48, 32), 128, 0, stream>>>(xh, wqt, wkt, wvt, tpos, Qh, Kt, Vh);
    attn_kernel<<<dim3(1024), 128, 0, stream>>>(Qh, Kt, Vh, ctxh);
    proj_kernel<<<dim3(16, 32), 128, 0, stream>>>(ctxh, wot, out);
}
